// TinyGPT_67147518705850
// MI455X (gfx1250) — matrix operand order repaired and checked by reading, not compiled, not run on hardware
//
#include <hip/hip_runtime.h>

// ---------------- model constants ----------------
#define BB      2
#define TT      4096
#define DMODEL  512
#define NHEADS  8
#define DHEAD   64
#define DFF     2048
#define NDEPTH  4
#define OUTD    64
#define MROWS   (BB*TT)        // 8192 token rows

typedef __attribute__((ext_vector_type(16))) _Float16 v16h;
typedef __attribute__((ext_vector_type(8)))  _Float16 v8h;
typedef __attribute__((ext_vector_type(8)))  float    v8f;

__device__ __forceinline__ v8f wmma16(v16h a, v16h b, v8f c) {
  // D = A(16x32 f16) * B(32x16 f16) + C(16x16 f32)
  return __builtin_amdgcn_wmma_f32_16x16x32_f16(false, a, false, b, (short)0, c,
                                                false, false);
}

__device__ __forceinline__ v16h cat8(v8h lo, v8h hi) {
  v16h r;
#pragma unroll
  for (int i = 0; i < 8; ++i) { r[i] = lo[i]; r[i + 8] = hi[i]; }
  return r;
}

// ---------------- DPP16 butterfly reductions over 16-lane halves ----------------
// xor1 = quad_perm[1,0,3,2] (0xB1), xor2 = quad_perm[2,3,0,1] (0x4E),
// row_half_mirror (0x141) == xor4 once quads are uniform,
// row_mirror (0x140) == xor8 once groups of 8 are uniform. Pure VALU, no LDS.
#if __has_builtin(__builtin_amdgcn_update_dpp)
template <int CTRL>
__device__ __forceinline__ float dppf(float x) {
  return __int_as_float(
      __builtin_amdgcn_update_dpp(0, __float_as_int(x), CTRL, 0xf, 0xf, true));
}
__device__ __forceinline__ float red16_max(float x) {
  x = fmaxf(x, dppf<0xB1>(x));
  x = fmaxf(x, dppf<0x4E>(x));
  x = fmaxf(x, dppf<0x141>(x));
  x = fmaxf(x, dppf<0x140>(x));
  return x;
}
__device__ __forceinline__ float red16_sum(float x) {
  x += dppf<0xB1>(x);
  x += dppf<0x4E>(x);
  x += dppf<0x141>(x);
  x += dppf<0x140>(x);
  return x;
}
#else
__device__ __forceinline__ float red16_max(float x) {
#pragma unroll
  for (int off = 1; off < 16; off <<= 1) x = fmaxf(x, __shfl_xor(x, off, 32));
  return x;
}
__device__ __forceinline__ float red16_sum(float x) {
#pragma unroll
  for (int off = 1; off < 16; off <<= 1) x += __shfl_xor(x, off, 32);
  return x;
}
#endif

// ---------------- fp32 -> fp16 weight conversion ----------------
__global__ void cvt_f16_kernel(const float* __restrict__ s, _Float16* __restrict__ d, int n) {
  int i = blockIdx.x * blockDim.x + threadIdx.x;
  if (i < n) d[i] = (_Float16)s[i];
}

// ---------------- embedding: x = x_in * inp_w + inp_b + pos ----------------
__global__ void embed_kernel(const float* __restrict__ x_in, const float* __restrict__ inp_w,
                             const float* __restrict__ inp_b, const float* __restrict__ pos,
                             float* __restrict__ x) {
  int idx = blockIdx.x * blockDim.x + threadIdx.x;      // over MROWS*DMODEL
  if (idx >= MROWS * DMODEL) return;
  int d  = idx & (DMODEL - 1);
  int bt = idx >> 9;                                    // DMODEL == 512
  int t  = bt & (TT - 1);
  x[idx] = x_in[bt] * inp_w[d] + inp_b[d] + pos[t * DMODEL + d];
}

// ---------------- layernorm: f32 in -> f16 out ----------------
__global__ __launch_bounds__(256)
void layernorm_kernel(const float* __restrict__ x, const float* __restrict__ g,
                      const float* __restrict__ bt, _Float16* __restrict__ out) {
  const int row = blockIdx.x, tid = threadIdx.x;
  const int wid = tid >> 5, lane = tid & 31;
  const float2 v = ((const float2*)(x + (size_t)row * DMODEL))[tid];
  __shared__ float red[8];

  float s = v.x + v.y;
#pragma unroll
  for (int off = 1; off < 32; off <<= 1) s += __shfl_xor(s, off, 32);
  if (lane == 0) red[wid] = s;
  __syncthreads();
  float tot = 0.f;
#pragma unroll
  for (int i = 0; i < 8; ++i) tot += red[i];
  const float mu = tot * (1.0f / DMODEL);

  const float d0 = v.x - mu, d1 = v.y - mu;
  float q = d0 * d0 + d1 * d1;
#pragma unroll
  for (int off = 1; off < 32; off <<= 1) q += __shfl_xor(q, off, 32);
  __syncthreads();
  if (lane == 0) red[wid] = q;
  __syncthreads();
  float vt = 0.f;
#pragma unroll
  for (int i = 0; i < 8; ++i) vt += red[i];
  const float rstd = rsqrtf(vt * (1.0f / DMODEL) + 1e-5f);

  const int c = 2 * tid;
  out[(size_t)row * DMODEL + c]     = (_Float16)(d0 * rstd * g[c]     + bt[c]);
  out[(size_t)row * DMODEL + c + 1] = (_Float16)(d1 * rstd * g[c + 1] + bt[c + 1]);
}

// ---------------- WMMA GEMM: C[M,N] = A[M,K](f16) * B[K,N](f16) ----------------
// Block tile 128x64 (8 waves, 32x32 per wave, 4 WMMA/step), LDS double-buffered.
// MODE 0: f16 out | 1: f16 out + exact GELU | 2: f32 residual += | 3: f32 out + bias
template <int MODE>
__global__ __launch_bounds__(256)
void gemm_wmma_kernel(const _Float16* __restrict__ A, const _Float16* __restrict__ Bw,
                      float* __restrict__ Cf, _Float16* __restrict__ Ch,
                      const float* __restrict__ bias, int N, int K) {
  __shared__ _Float16 sA[2][128 * 32];   // [buf][row][k]
  __shared__ _Float16 sBt[2][64 * 40];   // [buf][n][k], 40-half pitch (16B-aligned rows)
  const int tid  = threadIdx.x;
  const int wid  = tid >> 5, lane = tid & 31;
  const int bm   = blockIdx.y * 128, bn = blockIdx.x * 64;
  const int wm   = (wid >> 1) * 32;      // wave row offset (4 waves in M)
  const int wn   = (wid & 1) * 32;       // wave col offset (2 waves in N)
  const int col  = lane & 15, hlf = lane >> 4;
  const int ar   = tid >> 1, ac = (tid & 1) * 16;  // A staging: 128 rows x 32k
  const int br   = tid >> 3, bc = (tid & 7) * 8;   // B staging: 32 rows x 64n

  const _Float16* Arow = A  + (size_t)(bm + ar) * K + ac;
  const _Float16* Brow = Bw + (size_t)br * N + (bn + bc);

  // prologue: stage K-tile 0 into buffer 0
  {
    v8h a0 = *(const v8h*)(Arow);
    v8h a1 = *(const v8h*)(Arow + 8);
    *(v8h*)(&sA[0][ar * 32 + ac])     = a0;
    *(v8h*)(&sA[0][ar * 32 + ac + 8]) = a1;
    v8h bv = *(const v8h*)(Brow);
#pragma unroll
    for (int j = 0; j < 8; ++j) sBt[0][(bc + j) * 40 + br] = bv[j];
  }

  v8f acc00 = {}, acc01 = {}, acc10 = {}, acc11 = {};
  const int nk = K >> 5;
  int cur = 0;
  for (int kt = 0; kt < nk; ++kt) {
    __syncthreads();                      // buf[cur] staged; buf[cur^1] reads done
    const int nxt = cur ^ 1;
    const bool more = (kt + 1 < nk);
    v8h a0n = {}, a1n = {}, bvn = {};
    if (more) {                           // issue next-tile global loads early
      const _Float16* ap = Arow + (size_t)(kt + 1) * 32;
      a0n = *(const v8h*)ap;
      a1n = *(const v8h*)(ap + 8);
      bvn = *(const v8h*)(Brow + (size_t)(kt + 1) * 32 * N);
    }
    if (kt + 2 < nk) {                    // L2 prefetch two tiles ahead
      __builtin_prefetch(Arow + (size_t)(kt + 2) * 32, 0, 1);
      __builtin_prefetch(Brow + (size_t)(kt + 2) * 32 * N, 0, 1);
    }

    // B fragments: lane col = n, K = hlf*16 + e (contiguous)
    const v8h* pb0 = (const v8h*)&sBt[cur][(wn + col) * 40 + hlf * 16];
    v16h bf0 = cat8(pb0[0], pb0[1]);
    const v8h* pb1 = (const v8h*)&sBt[cur][(wn + 16 + col) * 40 + hlf * 16];
    v16h bf1 = cat8(pb1[0], pb1[1]);
    // A fragments: lane row = col, K = hlf*16 + e (contiguous)
    const v8h* pa0 = (const v8h*)&sA[cur][(wm + col) * 32 + hlf * 16];
    v16h af0 = cat8(pa0[0], pa0[1]);
    const v8h* pa1 = (const v8h*)&sA[cur][(wm + 16 + col) * 32 + hlf * 16];
    v16h af1 = cat8(pa1[0], pa1[1]);

    acc00 = wmma16(af0, bf0, acc00);
    acc01 = wmma16(af0, bf1, acc01);
    acc10 = wmma16(af1, bf0, acc10);
    acc11 = wmma16(af1, bf1, acc11);

    if (more) {                           // stage next tile into the other buffer
      *(v8h*)(&sA[nxt][ar * 32 + ac])     = a0n;
      *(v8h*)(&sA[nxt][ar * 32 + ac + 8]) = a1n;
#pragma unroll
      for (int j = 0; j < 8; ++j) sBt[nxt][(bc + j) * 40 + br] = bvn[j];
      cur = nxt;
    }
  }

  const int rb = hlf * 8;
#pragma unroll
  for (int tm = 0; tm < 2; ++tm) {
#pragma unroll
    for (int tn = 0; tn < 2; ++tn) {
      v8f acc = (tm == 0) ? (tn == 0 ? acc00 : acc01) : (tn == 0 ? acc10 : acc11);
      const int rowbase = bm + wm + tm * 16 + rb;   // C: VGPR r -> row r + 8*hlf
      const int cn = bn + wn + tn * 16 + col;
#pragma unroll
      for (int r = 0; r < 8; ++r) {
        const size_t o = (size_t)(rowbase + r) * N + cn;
        float v = acc[r];
        if (MODE == 0) {
          Ch[o] = (_Float16)v;
        } else if (MODE == 1) {
          float ge = 0.5f * v * (1.0f + erff(v * 0.70710678118654752f));
          Ch[o] = (_Float16)ge;
        } else if (MODE == 2) {
          Cf[o] += v;
        } else {
          Cf[o] = v + bias[cn];
        }
      }
    }
  }
}

// ---------------- flash attention (causal), WMMA for QK^T and P*V ----------------
// qkv: f16 [B, T, 3*DMODEL] (q | k | v). out: f16 [B, T, DMODEL].
__global__ __launch_bounds__(128)
void attn_kernel(const _Float16* __restrict__ qkv, _Float16* __restrict__ out) {
  __shared__ _Float16 sVt[64 * 40];       // V tile transposed: [d][k], padded pitch
  __shared__ _Float16 sP[4][16 * 32];     // per-wave P (C-layout -> A-layout bounce)
  const int tid  = threadIdx.x, wid = tid >> 5, lane = tid & 31;
  const int q0   = blockIdx.x * 64;
  const int h    = blockIdx.y;
  const int b    = blockIdx.z;
  const int qbase = q0 + wid * 16;
  const int col  = lane & 15, hlf = lane >> 4;
  const int RS   = 3 * DMODEL;            // 1536 halves per token row
  const _Float16* qkvb = qkv + (size_t)b * TT * RS;

  // Q fragments, pre-scaled by 1/sqrt(DHEAD) = 0.125 (exact power of two in f16)
  const _Float16* qptr = qkvb + (size_t)(qbase + col) * RS + h * DHEAD;
  v16h qa0 = cat8(*(const v8h*)(qptr + hlf * 16),      *(const v8h*)(qptr + 8 + hlf * 16));
  v16h qa1 = cat8(*(const v8h*)(qptr + 32 + hlf * 16), *(const v8h*)(qptr + 40 + hlf * 16));
#pragma unroll
  for (int i = 0; i < 16; ++i) {
    qa0[i] = qa0[i] * (_Float16)0.125f;
    qa1[i] = qa1[i] * (_Float16)0.125f;
  }

  float m[8], lsum[8];
  v8f oacc[4] = {{}, {}, {}, {}};
#pragma unroll
  for (int r = 0; r < 8; ++r) { m[r] = -3.0e38f; lsum[r] = 0.f; }

  const int kend = q0 + 64;
  for (int kb = 0; kb < kend; kb += 32) {
    // ---- stage V tile (32 k x 64 d) transposed into LDS (block-shared) ----
    {
      int kr  = tid >> 2;            // 0..31
      int dch = (tid & 3) * 16;      // 0,16,32,48
      const _Float16* vp = qkvb + (size_t)(kb + kr) * RS + 2 * DMODEL + h * DHEAD + dch;
      v8h a0 = *(const v8h*)vp;
      v8h a1 = *(const v8h*)(vp + 8);
#pragma unroll
      for (int j = 0; j < 8; ++j) {
        sVt[(dch + j) * 40 + kr]     = a0[j];
        sVt[(dch + 8 + j) * 40 + kr] = a1[j];
      }
    }
    __syncthreads();

    // ---- S = Q K^T for two 16-key tiles (K frags straight from global) ----
    v8f s0 = {}, s1 = {};
    {
      const _Float16* kp = qkvb + (size_t)(kb + col) * RS + DMODEL + h * DHEAD + hlf * 16;
      v16h kf0 = cat8(*(const v8h*)kp,        *(const v8h*)(kp + 8));
      v16h kf1 = cat8(*(const v8h*)(kp + 32), *(const v8h*)(kp + 40));
      s0 = wmma16(qa0, kf0, s0);
      s0 = wmma16(qa1, kf1, s0);
      const _Float16* kp2 = kp + (size_t)16 * RS;
      v16h kg0 = cat8(*(const v8h*)kp2,        *(const v8h*)(kp2 + 8));
      v16h kg1 = cat8(*(const v8h*)(kp2 + 32), *(const v8h*)(kp2 + 40));
      s1 = wmma16(qa0, kg0, s1);
      s1 = wmma16(qa1, kg1, s1);
    }

    // ---- causal mask + online softmax (DPP16 butterflies, no LDS) ----
    const int rbase = hlf * 8;
    float p0[8], p1[8];
#pragma unroll
    for (int r = 0; r < 8; ++r) {
      const int qg = qbase + rbase + r;
      float v0 = (kb + col      <= qg) ? s0[r] : -3.0e38f;
      float v1 = (kb + 16 + col <= qg) ? s1[r] : -3.0e38f;
      float mx = red16_max(fmaxf(v0, v1));
      const float mnew = fmaxf(m[r], mx);
      const float corr = __expf(m[r] - mnew);
      const float e0 = __expf(v0 - mnew);
      const float e1 = __expf(v1 - mnew);
      const float rs = red16_sum(e0 + e1);
      lsum[r] = lsum[r] * corr + rs;
      m[r] = mnew;
#pragma unroll
      for (int j = 0; j < 4; ++j) oacc[j][r] *= corr;
      p0[r] = e0; p1[r] = e1;
    }

    // ---- P: C-layout -> A-layout via per-wave LDS bounce ----
    _Float16* pw = &sP[wid][0];
#pragma unroll
    for (int r = 0; r < 8; ++r) {
      const int prow = rbase + r;
      pw[prow * 32 + col]      = (_Float16)p0[r];
      pw[prow * 32 + 16 + col] = (_Float16)p1[r];
    }
    __builtin_amdgcn_wave_barrier();   // DS ops are in-order within a wave

    const _Float16* pr = &sP[wid][col * 32];
    v16h pa = cat8(*(const v8h*)(pr + hlf * 16), *(const v8h*)(pr + 8 + hlf * 16));

    // ---- O += P * V  (4 d-tiles, shared A fragment) ----
#pragma unroll
    for (int j = 0; j < 4; ++j) {
      const _Float16* vb = &sVt[(j * 16 + col) * 40 + hlf * 16];
      v16h vf = cat8(*(const v8h*)vb, *(const v8h*)(vb + 8));
      oacc[j] = wmma16(pa, vf, oacc[j]);
    }
    __syncthreads();   // protect sVt before next iteration's staging
  }

  // ---- normalize and write ----
#pragma unroll
  for (int r = 0; r < 8; ++r) {
    const int qg = qbase + hlf * 8 + r;
    const float inv = 1.0f / lsum[r];
    const size_t ob = ((size_t)b * TT + qg) * DMODEL + h * DHEAD;
#pragma unroll
    for (int j = 0; j < 4; ++j)
      out[ob + j * 16 + col] = (_Float16)(oacc[j][r] * inv);
  }
}

// ---------------- host launcher ----------------
extern "C" void kernel_launch(void* const* d_in, const int* in_sizes, int n_in,
                              void* d_out, int out_size, void* d_ws, size_t ws_size,
                              hipStream_t stream) {
  const float* x_in   = (const float*)d_in[0];
  const float* inp_w  = (const float*)d_in[1];
  const float* inp_b  = (const float*)d_in[2];
  const float* pos    = (const float*)d_in[3];
  const float* ln1_g  = (const float*)d_in[4];
  const float* ln1_b  = (const float*)d_in[5];
  const float* qkv_w  = (const float*)d_in[6];
  const float* proj_w = (const float*)d_in[7];
  const float* ln2_g  = (const float*)d_in[8];
  const float* ln2_b  = (const float*)d_in[9];
  const float* fc1_w  = (const float*)d_in[10];
  const float* fc2_w  = (const float*)d_in[11];
  const float* lnf_g  = (const float*)d_in[12];
  const float* lnf_b  = (const float*)d_in[13];
  const float* head_w = (const float*)d_in[14];
  const float* head_b = (const float*)d_in[15];
  float* out = (float*)d_out;

  char* ws = (char*)d_ws;
  auto take = [&](size_t bytes) { char* p = ws; ws += (bytes + 255) & ~(size_t)255; return p; };
  float*    xbuf  = (float*)   take((size_t)MROWS * DMODEL * 4);
  _Float16* lnbuf = (_Float16*)take((size_t)MROWS * DMODEL * 2);
  _Float16* qkvb  = (_Float16*)take((size_t)MROWS * 3 * DMODEL * 2);
  _Float16* attnb = (_Float16*)take((size_t)MROWS * DMODEL * 2);
  _Float16* ffb   = (_Float16*)take((size_t)MROWS * DFF * 2);
  _Float16* wqkv  = (_Float16*)take((size_t)NDEPTH * DMODEL * 3 * DMODEL * 2);
  _Float16* wproj = (_Float16*)take((size_t)NDEPTH * DMODEL * DMODEL * 2);
  _Float16* wfc1  = (_Float16*)take((size_t)NDEPTH * DMODEL * DFF * 2);
  _Float16* wfc2  = (_Float16*)take((size_t)NDEPTH * DFF * DMODEL * 2);
  _Float16* whead = (_Float16*)take((size_t)DMODEL * OUTD * 2);

  auto cvt = [&](const float* s, _Float16* d, int n) {
    cvt_f16_kernel<<<(n + 255) / 256, 256, 0, stream>>>(s, d, n);
  };
  cvt(qkv_w,  wqkv,  NDEPTH * DMODEL * 3 * DMODEL);
  cvt(proj_w, wproj, NDEPTH * DMODEL * DMODEL);
  cvt(fc1_w,  wfc1,  NDEPTH * DMODEL * DFF);
  cvt(fc2_w,  wfc2,  NDEPTH * DFF * DMODEL);
  cvt(head_w, whead, DMODEL * OUTD);

  {
    int n = MROWS * DMODEL;
    embed_kernel<<<(n + 255) / 256, 256, 0, stream>>>(x_in, inp_w, inp_b, pos, xbuf);
  }

  const dim3 gemmBlk(256);
  for (int l = 0; l < NDEPTH; ++l) {
    layernorm_kernel<<<MROWS, 256, 0, stream>>>(xbuf, ln1_g + l * DMODEL, ln1_b + l * DMODEL, lnbuf);
    gemm_wmma_kernel<0><<<dim3((3 * DMODEL) / 64, MROWS / 128), gemmBlk, 0, stream>>>(
        lnbuf, wqkv + (size_t)l * DMODEL * 3 * DMODEL, nullptr, qkvb, nullptr, 3 * DMODEL, DMODEL);
    attn_kernel<<<dim3(TT / 64, NHEADS, BB), 128, 0, stream>>>(qkvb, attnb);
    gemm_wmma_kernel<2><<<dim3(DMODEL / 64, MROWS / 128), gemmBlk, 0, stream>>>(
        attnb, wproj + (size_t)l * DMODEL * DMODEL, xbuf, nullptr, nullptr, DMODEL, DMODEL);

    layernorm_kernel<<<MROWS, 256, 0, stream>>>(xbuf, ln2_g + l * DMODEL, ln2_b + l * DMODEL, lnbuf);
    gemm_wmma_kernel<1><<<dim3(DFF / 64, MROWS / 128), gemmBlk, 0, stream>>>(
        lnbuf, wfc1 + (size_t)l * DMODEL * DFF, nullptr, ffb, nullptr, DFF, DMODEL);
    gemm_wmma_kernel<2><<<dim3(DMODEL / 64, MROWS / 128), gemmBlk, 0, stream>>>(
        ffb, wfc2 + (size_t)l * DFF * DMODEL, xbuf, nullptr, nullptr, DMODEL, DFF);
  }

  layernorm_kernel<<<MROWS, 256, 0, stream>>>(xbuf, lnf_g, lnf_b, lnbuf);
  gemm_wmma_kernel<3><<<dim3(OUTD / 64, MROWS / 128), gemmBlk, 0, stream>>>(
      lnbuf, whead, out, nullptr, head_b, OUTD, DMODEL);
}
